// QLSTM_65481071403835
// MI455X (gfx1250) — matrix operand order repaired and checked by reading, not compiled, not run on hardware
//
#include <hip/hip_runtime.h>

typedef __attribute__((ext_vector_type(16))) _Float16 v16h;
typedef __attribute__((ext_vector_type(8)))  _Float16 v8h;
typedef __attribute__((ext_vector_type(8)))  float    v8f;

#define TSTEPS 128
#define BATCH  4096
#define DIN    128
#define HID    8
#define FANIN  136
#define NROW   32   // 4 gates * 8 hidden

// ---------------------------------------------------------------------------
// Prep: pack the 4 gate weight matrices (8 x 136 f32 each) into a combined
// 32 x 136 f16 matrix (row n = gate*8 + j), and zero the h/c recurrent state.
// Runs every launch (harness poisons buffers once; replays must be identical).
// ---------------------------------------------------------------------------
__global__ __launch_bounds__(256) void qlstm_prep(
    const float* __restrict__ Wf, const float* __restrict__ Wi,
    const float* __restrict__ Wu, const float* __restrict__ Wo,
    _Float16* __restrict__ wpack, float* __restrict__ h, float* __restrict__ c)
{
    int i = blockIdx.x * blockDim.x + threadIdx.x;
    if (i < NROW * FANIN) {
        int n = i / FANIN, k = i - n * FANIN;
        const float* W = (n < 8) ? Wf : (n < 16) ? Wi : (n < 24) ? Wu : Wo;
        wpack[i] = (_Float16)W[(n & 7) * FANIN + k];
    }
    if (i < BATCH * HID) { h[i] = 0.f; c[i] = 0.f; }
}

// ---------------------------------------------------------------------------
// Input projection GEMM with WMMA: xproj[T*B, 32] = x[T*B,128] @ Wx^T.
// One wave per 16-row M tile; two 16-col N tiles; K unrolled 128/32 = 4.
// A fragment (16x32 f16, ISA 7.12.2): lane l<16 row=l holds K {k0..k0+7,
// k0+16..k0+23}; lane l>=16 row=l-16 holds K {k0+8..+15, k0+24..+31}.
// B fragment (32x16 f16): lane holds col N=lane&15, K {k0+(lane>>4)*8..+7,
// k0+16+(lane>>4)*8..+7} (a row of W, since B = W^T), same order as A.
// D (16x16 f32): acc[k] -> row m0 + k + 8*(lane>>4), col lane&15.
// ---------------------------------------------------------------------------
__global__ __launch_bounds__(256) void qlstm_xproj(
    const float* __restrict__ x, const _Float16* __restrict__ wpack,
    float* __restrict__ xproj)
{
    int wid  = (blockIdx.x * blockDim.x + threadIdx.x) >> 5;
    int lane = threadIdx.x & 31;
    int m0   = wid * 16;
    int lhi  = lane >> 4;    // 0/1
    int llo  = lane & 15;

    v8f acc0 = {}; v8f acc1 = {};
    const float* arow = x + (size_t)(m0 + llo) * DIN + lhi * 8;

#pragma unroll
    for (int k0 = 0; k0 < DIN; k0 += 32) {
        // ---- A fragment: convert f32 -> f16 in registers ----
        const float4* ap = (const float4*)(arow + k0);
        float4 f0 = ap[0], f1 = ap[1];   // K chunk base .. base+7
        float4 f2 = ap[4], f3 = ap[5];   // K chunk base+16 .. base+23
        v16h a;
        a[0]=(_Float16)f0.x;  a[1]=(_Float16)f0.y;  a[2]=(_Float16)f0.z;  a[3]=(_Float16)f0.w;
        a[4]=(_Float16)f1.x;  a[5]=(_Float16)f1.y;  a[6]=(_Float16)f1.z;  a[7]=(_Float16)f1.w;
        a[8]=(_Float16)f2.x;  a[9]=(_Float16)f2.y;  a[10]=(_Float16)f2.z; a[11]=(_Float16)f2.w;
        a[12]=(_Float16)f3.x; a[13]=(_Float16)f3.y; a[14]=(_Float16)f3.z; a[15]=(_Float16)f3.w;

        // ---- B fragments (already f16 in wpack) ----
        const _Float16* b0p = wpack + (size_t)llo * FANIN        + k0 + lhi * 8;
        const _Float16* b1p = wpack + (size_t)(16 + llo) * FANIN + k0 + lhi * 8;
        v8h b0lo = *(const v8h*)b0p, b0hi = *(const v8h*)(b0p + 16);
        v8h b1lo = *(const v8h*)b1p, b1hi = *(const v8h*)(b1p + 16);
        v16h b0, b1;
#pragma unroll
        for (int i = 0; i < 8; ++i) {
            b0[i] = b0lo[i]; b0[i + 8] = b0hi[i];
            b1[i] = b1lo[i]; b1[i + 8] = b1hi[i];
        }

        acc0 = __builtin_amdgcn_wmma_f32_16x16x32_f16(false, a, false, b0,
                                                      (short)0, acc0, false, false);
        acc1 = __builtin_amdgcn_wmma_f32_16x16x32_f16(false, a, false, b1,
                                                      (short)0, acc1, false, false);
    }

#pragma unroll
    for (int k = 0; k < 8; ++k) {
        size_t row = (size_t)(m0 + k + lhi * 8);
        xproj[row * NROW + llo]      = acc0[k];
        xproj[row * NROW + 16 + llo] = acc1[k];
    }
}

// ---------------------------------------------------------------------------
// 8-qubit statevector simulation for one gate: one wave, 8 complex amps/lane.
// idx = (lane<<3) | j ; qubit q lives at bit (7-q). Cross-lane qubits (q<5)
// pair via shfl_xor with mask 16>>q; intra-lane qubits (q=5,6,7) pair within
// the register array. Returns raw <Z_{lane&7}> expectation (same in all lanes
// of each group of 8).
// ---------------------------------------------------------------------------
__device__ __forceinline__ float qsim_gate(
    const float* __restrict__ xp8,        // xproj row base for this gate (8 vals)
    const float* __restrict__ W,          // gate weight (8 x 136), recurrent cols 128..135
    const float* __restrict__ bias,       // (8,)
    const float* __restrict__ theta,      // (2, 8)
    const float* hk, int lane)
{
    int q = lane & 7;
    // pre-activation RX angle for qubit q
    float pre = xp8[q] + bias[q];
#pragma unroll
    for (int k = 0; k < 8; ++k) pre += hk[k] * W[q * FANIN + DIN + k];
    float ha = 0.5f * pre;
    float cl = __cosf(ha), sl = __sinf(ha);

    float cq[8], sq[8];
#pragma unroll
    for (int qq = 0; qq < 8; ++qq) {
        cq[qq] = __shfl(cl, qq, 32);
        sq[qq] = __shfl(sl, qq, 32);
    }

    // initial product state: magnitude * (-i)^popcount(idx)
    float mhigh = 1.f;
#pragma unroll
    for (int qq = 0; qq < 5; ++qq) {
        int bit = (lane >> (4 - qq)) & 1;
        mhigh *= bit ? sq[qq] : cq[qq];
    }
    float re[8], im[8];
#pragma unroll
    for (int j = 0; j < 8; ++j) {
        float m = mhigh;
        m *= ((j >> 2) & 1) ? sq[5] : cq[5];
        m *= ((j >> 1) & 1) ? sq[6] : cq[6];
        m *= (j & 1)        ? sq[7] : cq[7];
        int ph = (__popc(lane) + __popc(j)) & 3;   // (-i)^ph
        re[j] = (ph == 0) ? m : (ph == 2) ? -m : 0.f;
        im[j] = (ph == 1) ? -m : (ph == 3) ? m : 0.f;
    }

    // DEPTH layers of RY(theta) per qubit followed by CZ chain
#pragma unroll
    for (int d = 0; d < 2; ++d) {
#pragma unroll
        for (int qq = 0; qq < 8; ++qq) {
            float th = 0.5f * theta[d * 8 + qq];
            float c = __cosf(th), s = __sinf(th);
            if (qq < 5) {
                int mask = 16 >> qq;
                int bit  = (lane >> (4 - qq)) & 1;
                float sg = bit ? s : -s;           // new = c*self + (+/-s)*partner
#pragma unroll
                for (int j = 0; j < 8; ++j) {
                    float pr = __shfl_xor(re[j], mask, 32);
                    float pi = __shfl_xor(im[j], mask, 32);
                    re[j] = c * re[j] + sg * pr;
                    im[j] = c * im[j] + sg * pi;
                }
            } else {
                int st = 1 << (7 - qq);            // 4,2,1
#pragma unroll
                for (int j = 0; j < 8; ++j) {
                    if ((j & st) == 0) {
                        int jp = j | st;
                        float ar = re[j], ai = im[j];
                        float br = re[jp], bi = im[jp];
                        re[j]  = c * ar - s * br;  im[j]  = c * ai - s * bi;
                        re[jp] = s * ar + c * br;  im[jp] = s * ai + c * bi;
                    }
                }
            }
        }
        // CZ chain: sign = (-1)^popc(idx & (idx>>1))
#pragma unroll
        for (int j = 0; j < 8; ++j) {
            int idx = (lane << 3) | j;
            if (__popc(idx & (idx >> 1)) & 1) { re[j] = -re[j]; im[j] = -im[j]; }
        }
    }

    // measurement: <Z_q> = sum_idx p_idx * (1 - 2*bit_q)
    float pz[8];
#pragma unroll
    for (int qq = 0; qq < 8; ++qq) pz[qq] = 0.f;
#pragma unroll
    for (int j = 0; j < 8; ++j) {
        int idx = (lane << 3) | j;
        float p = re[j] * re[j] + im[j] * im[j];
#pragma unroll
        for (int qq = 0; qq < 8; ++qq)
            pz[qq] += ((idx >> (7 - qq)) & 1) ? -p : p;
    }
    float act = 0.f;
#pragma unroll
    for (int qq = 0; qq < 8; ++qq) {
#pragma unroll
        for (int off = 16; off; off >>= 1)
            pz[qq] += __shfl_xor(pz[qq], off, 32);
        act += (q == qq) ? pz[qq] : 0.f;
    }
    return act;
}

// ---------------------------------------------------------------------------
// One LSTM time step: one wave per (batch, gate) for 4x the per-step
// parallelism (16384 waves/step); 8 waves (2 batch elements) per block.
// The four gate activations of a batch element meet in LDS; one barrier,
// then the g==0 wave performs the LSTM cell update.
// ---------------------------------------------------------------------------
__global__ __launch_bounds__(256) void qlstm_step(
    const float* __restrict__ xproj,
    const float* __restrict__ Wf, const float* __restrict__ bf, const float* __restrict__ thf,
    const float* __restrict__ Wi, const float* __restrict__ bi, const float* __restrict__ thi,
    const float* __restrict__ Wu, const float* __restrict__ bu, const float* __restrict__ thu,
    const float* __restrict__ Wo, const float* __restrict__ bo, const float* __restrict__ tho,
    float* __restrict__ h, float* __restrict__ c, float* __restrict__ out, int t)
{
    __shared__ float acts[2][4][HID];

    int wv   = threadIdx.x >> 5;     // 0..7
    int lane = threadIdx.x & 31;
    int sub  = wv >> 2;              // batch element within block (0/1)
    int g    = wv & 3;               // gate: 0=f 1=i 2=u 3=o (wave-uniform)
    int b    = blockIdx.x * 2 + sub;

    float hv = (lane < 8) ? h[b * 8 + lane] : 0.f;
    float hk[8];
#pragma unroll
    for (int k = 0; k < 8; ++k) hk[k] = __shfl(hv, k, 32);

    const float* Wg[4]  = { Wf, Wi, Wu, Wo };
    const float* bg[4]  = { bf, bi, bu, bo };
    const float* thg[4] = { thf, thi, thu, tho };

    const float* xb = xproj + ((size_t)t * BATCH + b) * NROW + g * HID;
    float act = qsim_gate(xb, Wg[g], bg[g], thg[g], hk, lane);

    // gate nonlinearity: u -> tanh, others -> sigmoid
    float val = (g == 2) ? tanhf(act) : 1.f / (1.f + __expf(-act));
    if (lane < 8) acts[sub][g][lane] = val;
    __syncthreads();

    if (g == 0 && lane < 8) {
        float fg = acts[sub][0][lane];
        float ig = acts[sub][1][lane];
        float gg = acts[sub][2][lane];
        float og = acts[sub][3][lane];
        float cp = c[b * 8 + lane];
        float cn = fg * cp + ig * gg;
        float hn = og * tanhf(cn);
        c[b * 8 + lane] = cn;
        h[b * 8 + lane] = hn;
        out[((size_t)t * BATCH + b) * 8 + lane] = hn;
        if (t == TSTEPS - 1) {
            size_t tail = (size_t)TSTEPS * BATCH * 8;
            out[tail + (size_t)b * 8 + lane] = hn;                        // hx
            out[tail + (size_t)BATCH * 8 + (size_t)b * 8 + lane] = cn;    // cx
        }
    }
}

// ---------------------------------------------------------------------------
extern "C" void kernel_launch(void* const* d_in, const int* in_sizes, int n_in,
                              void* d_out, int out_size, void* d_ws, size_t ws_size,
                              hipStream_t stream)
{
    const float* x   = (const float*)d_in[0];
    const float* Wf  = (const float*)d_in[1];
    const float* bf  = (const float*)d_in[2];
    const float* thf = (const float*)d_in[3];
    const float* Wi  = (const float*)d_in[4];
    const float* bi  = (const float*)d_in[5];
    const float* thi = (const float*)d_in[6];
    const float* Wu  = (const float*)d_in[7];
    const float* bu  = (const float*)d_in[8];
    const float* thu = (const float*)d_in[9];
    const float* Wo  = (const float*)d_in[10];
    const float* bo  = (const float*)d_in[11];
    const float* tho = (const float*)d_in[12];
    float* out = (float*)d_out;

    char* ws = (char*)d_ws;
    _Float16* wpack = (_Float16*)ws;                      // 32*136*2 B
    float* xproj = (float*)(ws + 16384);                  // T*B*32*4 = 64 MB
    float* h = (float*)(ws + 16384 + (size_t)TSTEPS * BATCH * NROW * 4);
    float* c = h + BATCH * HID;

    // prep: pack weights to f16, zero recurrent state (every launch)
    qlstm_prep<<<128, 256, 0, stream>>>(Wf, Wi, Wu, Wo, wpack, h, c);

    // hoisted input projection (WMMA): 32768 M-tiles, 8 waves/block
    qlstm_xproj<<<4096, 256, 0, stream>>>(x, wpack, xproj);

    // sequential recurrence: 16384 waves/step, 2 batch elements per block
    for (int t = 0; t < TSTEPS; ++t)
        qlstm_step<<<BATCH / 2, 256, 0, stream>>>(xproj,
            Wf, bf, thf, Wi, bi, thi, Wu, bu, thu, Wo, bo, tho,
            h, c, out, t);
}
